// MPNN_46162308497548
// MI455X (gfx1250) — matrix operand order repaired and checked by reading, not compiled, not run on hardware
//
#include <hip/hip_runtime.h>
#include <hip/hip_bf16.h>

#define N_NODES 10000
#define N_EDGES 20000
#define N_GRAPHS 500
#define IN_CH 16
#define EC_CH 4
#define H1_CH 120
#define H2_CH 210
#define BN_EPS 1e-5f

typedef __bf16 v16bf __attribute__((ext_vector_type(16)));
typedef float  v8f   __attribute__((ext_vector_type(8)));

__device__ __forceinline__ void atomicAddF(float* p, float v) {
    __hip_atomic_fetch_add(p, v, __ATOMIC_RELAXED, __HIP_MEMORY_SCOPE_AGENT);
}

// t[e,k] = relu(ea @ wa + ba), k in [0,32)
__global__ void edge_mlp_kernel(const float* __restrict__ ea, const float* __restrict__ wa,
                                const float* __restrict__ ba, float* __restrict__ t) {
    int idx = blockIdx.x * blockDim.x + threadIdx.x;
    if (idx >= N_EDGES * 32) return;
    int e = idx >> 5, k = idx & 31;
    float v = ba[k];
#pragma unroll
    for (int c = 0; c < EC_CH; ++c) v += ea[e * EC_CH + c] * wa[c * 32 + k];
    t[idx] = fmaxf(v, 0.f);
}

// WT[o][K] bf16, K = k*INCH+i (k<32 -> wb, k==32 -> bias row), zero-padded.
template <int INCH, int OUTCH, int KPAD, int NPAD>
__global__ void prep_wT_kernel(const float* __restrict__ wb, const float* __restrict__ bb,
                               __bf16* __restrict__ WT) {
    int idx = blockIdx.x * blockDim.x + threadIdx.x;
    if (idx >= NPAD * KPAD) return;
    int o = idx / KPAD, K = idx % KPAD;
    float v = 0.f;
    if (o < OUTCH && K < 33 * INCH) {
        int k = K / INCH, i = K % INCH;
        v = (k < 32) ? wb[k * (INCH * OUTCH) + i * OUTCH + o] : bb[i * OUTCH + o];
    }
    WT[idx] = (__bf16)v;
}

__global__ void deg_kernel(const int* __restrict__ ei, float* __restrict__ cnt) {
    int e = blockIdx.x * blockDim.x + threadIdx.x;
    if (e < N_EDGES) atomicAddF(&cnt[ei[N_EDGES + e]], 1.f);
}

// Fused NNConv message GEMM:  msg = Z @ W'  with  Z[e, k*INCH+i] = t_ext[e,k]*x[src(e),i],
// scattered directly into aggr[dst] with atomics.
// Register blocking: MT=2 M-tiles (32 edges) x NTW=2 N-tiles per wave.
// A fragment built once per (M-tile, k-chunk) from LDS and reused across N tiles;
// B fragment loaded once per (N-tile, k-chunk) and reused across M tiles.
template <int INCH, int OUTCH, int KPAD, int NPAD, int MT, int NTW>
__global__ __launch_bounds__((NPAD / 16 / NTW) * 32) void nnconv_msg_wmma(
    const float* __restrict__ xin, const float* __restrict__ t,
    const __bf16* __restrict__ WT, const int* __restrict__ ei,
    float* __restrict__ aggr) {
    constexpr int KT = 33 * INCH;
    constexpr int NTILES = NPAD / 16;
    constexpr int WAVES = NTILES / NTW;
    constexpr int THREADS = WAVES * 32;
    constexpr int ME = MT * 16;                  // edges per block
    __shared__ float tl[ME][34];                 // t rows + implicit 1.0 bias term
    __shared__ float xs[ME][INCH];               // gathered x[src] rows
    __shared__ int dsts[ME];
    const int tid = threadIdx.x;
    const int e0 = blockIdx.x * ME;
    for (int idx = tid; idx < ME * 33; idx += THREADS) {
        int m = idx / 33, k = idx % 33;
        tl[m][k] = (k < 32) ? t[(e0 + m) * 32 + k] : 1.0f;
    }
    for (int idx = tid; idx < ME * INCH; idx += THREADS) {
        int m = idx / INCH, i = idx % INCH;
        xs[m][i] = xin[(size_t)ei[e0 + m] * INCH + i];
    }
    if (tid < ME) dsts[tid] = ei[N_EDGES + e0 + tid];
    __syncthreads();

    const int wave = tid >> 5, lane = tid & 31;
    const int mrow = lane & 15;                  // A row / B column within tile
    const int hi = lane >> 4;
    const int akoff = hi * 16;                   // contiguous A layout: upper lanes K+16
    const int bkoff = hi * 16;                   // contiguous B layout: upper lanes K+16

    const __bf16* brow[NTW];
#pragma unroll
    for (int w = 0; w < NTW; ++w)
        brow[w] = WT + (size_t)((wave + w * WAVES) * 16 + mrow) * KPAD + bkoff;

    v8f acc[MT][NTW];
#pragma unroll
    for (int mt = 0; mt < MT; ++mt)
#pragma unroll
        for (int w = 0; w < NTW; ++w) acc[mt][w] = (v8f){};

    for (int kc = 0; kc < KPAD / 32; ++kc) {
        const int kk = kc * 32;
#pragma unroll
        for (int w = 0; w < NTW; ++w)
            if (kk + 64 < KPAD) __builtin_prefetch(brow[w] + kk + 64, 0, 3);
        // Build MT A fragments (rank-1 Z tiles) from LDS, in contiguous k order.
        v16bf a[MT];
#pragma unroll
        for (int mt = 0; mt < MT; ++mt) {
#pragma unroll
            for (int j = 0; j < 16; ++j) {
                int K = kk + akoff + j;
                float v = (K < KT)
                        ? tl[mt * 16 + mrow][K / INCH] * xs[mt * 16 + mrow][K % INCH]
                        : 0.f;
                a[mt][j] = (__bf16)v;
            }
        }
        // NTW B fragments: 16 contiguous bf16 each (32B aligned).
        v16bf b[NTW];
#pragma unroll
        for (int w = 0; w < NTW; ++w) b[w] = *(const v16bf*)(brow[w] + kk);
        // MT x NTW WMMAs reuse the fragments.
#pragma unroll
        for (int w = 0; w < NTW; ++w)
#pragma unroll
            for (int mt = 0; mt < MT; ++mt)
                acc[mt][w] = __builtin_amdgcn_wmma_f32_16x16x32_bf16(
                    false, a[mt], false, b[w], (short)0, acc[mt][w], false, false);
    }

#pragma unroll
    for (int w = 0; w < NTW; ++w) {
        int n = (wave + w * WAVES) * 16 + mrow;
        if (n < OUTCH) {
#pragma unroll
            for (int mt = 0; mt < MT; ++mt)
#pragma unroll
                for (int r = 0; r < 8; ++r) {
                    int m = mt * 16 + r + hi * 8;   // ISA C/D layout
                    atomicAddF(&aggr[(size_t)dsts[m] * OUTCH + n], acc[mt][w][r]);
                }
        }
    }
}

// aggr/deg + x@root + bias, relu; accumulate BN sum/sumsq per channel.
template <int INCH, int OUTCH>
__global__ void combine_kernel(const float* __restrict__ aggr, const float* __restrict__ cnt,
                               const float* __restrict__ xin, const float* __restrict__ root,
                               const float* __restrict__ bias, float* __restrict__ pre,
                               float* __restrict__ bsum, float* __restrict__ bsq) {
    __shared__ float xr[INCH];
    int nn = blockIdx.x;
    for (int i = threadIdx.x; i < INCH; i += blockDim.x) xr[i] = xin[(size_t)nn * INCH + i];
    __syncthreads();
    int o = threadIdx.x;
    if (o >= OUTCH) return;
    float v = aggr[(size_t)nn * OUTCH + o] / fmaxf(cnt[nn], 1.f) + bias[o];
    for (int i = 0; i < INCH; ++i) v += xr[i] * root[i * OUTCH + o];
    v = fmaxf(v, 0.f);
    pre[(size_t)nn * OUTCH + o] = v;
    atomicAddF(&bsum[o], v);
    atomicAddF(&bsq[o], v * v);
}

template <int OUTCH>
__global__ void bn_apply_kernel(const float* __restrict__ pre, const float* __restrict__ bsum,
                                const float* __restrict__ bsq, const float* __restrict__ gamma,
                                const float* __restrict__ beta, float* __restrict__ h) {
    int idx = blockIdx.x * blockDim.x + threadIdx.x;
    if (idx >= N_NODES * OUTCH) return;
    int o = idx % OUTCH;
    float m = bsum[o] * (1.f / N_NODES);
    float var = bsq[o] * (1.f / N_NODES) - m * m;
    h[idx] = (pre[idx] - m) * rsqrtf(var + BN_EPS) * gamma[o] + beta[o];
}

__global__ void pool_kernel(const float* __restrict__ h2, const int* __restrict__ batch,
                            float* __restrict__ gsum, float* __restrict__ gcnt) {
    int idx = blockIdx.x * blockDim.x + threadIdx.x;
    if (idx >= N_NODES * H2_CH) return;
    int n = idx / H2_CH, o = idx % H2_CH;
    int g = batch[n];
    atomicAddF(&gsum[g * H2_CH + o], h2[idx]);
    if (o == 0) atomicAddF(&gcnt[g], 1.f);
}

__global__ void head_kernel(const float* __restrict__ gsum, const float* __restrict__ gcnt,
                            const float* __restrict__ w1, const float* __restrict__ b1,
                            const float* __restrict__ w2, const float* __restrict__ b2,
                            float* __restrict__ out) {
    __shared__ float gm[H2_CH];
    __shared__ float red[64];
    int g = blockIdx.x, t = threadIdx.x;
    float inv = 1.f / fmaxf(gcnt[g], 1.f);
    for (int c = t; c < H2_CH; c += 64) gm[c] = gsum[g * H2_CH + c] * inv;
    __syncthreads();
    float q = b1[t];
    for (int c = 0; c < H2_CH; ++c) q += gm[c] * w1[c * 64 + t];
    q = fmaxf(q, 0.f) * w2[t];
    red[t] = q;
    __syncthreads();
    for (int s = 32; s > 0; s >>= 1) {
        if (t < s) red[t] += red[t + s];
        __syncthreads();
    }
    if (t == 0) out[g] = red[0] + b2[0];
}

extern "C" void kernel_launch(void* const* d_in, const int* in_sizes, int n_in,
                              void* d_out, int out_size, void* d_ws, size_t ws_size,
                              hipStream_t stream) {
    (void)in_sizes; (void)n_in; (void)out_size; (void)ws_size;
    const float* x     = (const float*)d_in[0];
    const float* ea    = (const float*)d_in[1];
    const float* w1a   = (const float*)d_in[2];
    const float* b1a   = (const float*)d_in[3];
    const float* w1b   = (const float*)d_in[4];
    const float* b1b   = (const float*)d_in[5];
    const float* root1 = (const float*)d_in[6];
    const float* bias1 = (const float*)d_in[7];
    const float* gamma1= (const float*)d_in[8];
    const float* beta1 = (const float*)d_in[9];
    const float* w2a   = (const float*)d_in[10];
    const float* b2a   = (const float*)d_in[11];
    const float* w2b   = (const float*)d_in[12];
    const float* b2b   = (const float*)d_in[13];
    const float* root2 = (const float*)d_in[14];
    const float* bias2 = (const float*)d_in[15];
    const float* gamma2= (const float*)d_in[16];
    const float* beta2 = (const float*)d_in[17];
    const float* lin1w = (const float*)d_in[18];
    const float* lin1b = (const float*)d_in[19];
    const float* lin2w = (const float*)d_in[20];
    const float* lin2b = (const float*)d_in[21];
    const int*   ei    = (const int*)d_in[22];
    const int*   batch = (const int*)d_in[23];
    float* out = (float*)d_out;

    char* ws = (char*)d_ws;
    size_t off = 0;
    auto alloc = [&](size_t bytes) -> void* {
        void* p = ws + off;
        off = (off + bytes + 255) & ~(size_t)255;
        return p;
    };
    float*  t1   = (float*)alloc((size_t)N_EDGES * 32 * 4);
    float*  t2   = (float*)alloc((size_t)N_EDGES * 32 * 4);
    __bf16* W1T  = (__bf16*)alloc((size_t)128 * 544 * 2);
    __bf16* W2T  = (__bf16*)alloc((size_t)224 * 3968 * 2);
    float*  aggr = (float*)alloc((size_t)N_NODES * H2_CH * 4);
    float*  cnt  = (float*)alloc((size_t)N_NODES * 4);
    float*  pre  = (float*)alloc((size_t)N_NODES * H2_CH * 4);
    float*  h1   = (float*)alloc((size_t)N_NODES * H1_CH * 4);
    float*  h2   = (float*)alloc((size_t)N_NODES * H2_CH * 4);
    float*  bsum = (float*)alloc(256 * 4);
    float*  bsq  = (float*)alloc(256 * 4);
    float*  gsum = (float*)alloc((size_t)N_GRAPHS * H2_CH * 4);
    float*  gcnt = (float*)alloc((size_t)N_GRAPHS * 4);

    hipMemsetAsync(cnt, 0, (size_t)N_NODES * 4, stream);
    hipMemsetAsync(aggr, 0, (size_t)N_NODES * H2_CH * 4, stream);
    hipMemsetAsync(bsum, 0, 256 * 4, stream);
    hipMemsetAsync(bsq, 0, 256 * 4, stream);

    edge_mlp_kernel<<<(N_EDGES * 32 + 255) / 256, 256, 0, stream>>>(ea, w1a, b1a, t1);
    edge_mlp_kernel<<<(N_EDGES * 32 + 255) / 256, 256, 0, stream>>>(ea, w2a, b2a, t2);
    prep_wT_kernel<16, 120, 544, 128><<<(128 * 544 + 255) / 256, 256, 0, stream>>>(w1b, b1b, W1T);
    prep_wT_kernel<120, 210, 3968, 224><<<(224 * 3968 + 255) / 256, 256, 0, stream>>>(w2b, b2b, W2T);
    deg_kernel<<<(N_EDGES + 255) / 256, 256, 0, stream>>>(ei, cnt);

    // conv1: K = 33*16 = 528 (pad 544), N = 120 (pad 128); 4 waves, 32 edges/block
    nnconv_msg_wmma<16, 120, 544, 128, 2, 2>
        <<<N_EDGES / 32, 128, 0, stream>>>(x, t1, W1T, ei, aggr);
    combine_kernel<16, 120><<<N_NODES, 128, 0, stream>>>(aggr, cnt, x, root1, bias1, pre, bsum, bsq);
    bn_apply_kernel<120><<<(N_NODES * 120 + 255) / 256, 256, 0, stream>>>(pre, bsum, bsq, gamma1, beta1, h1);

    hipMemsetAsync(aggr, 0, (size_t)N_NODES * H2_CH * 4, stream);
    hipMemsetAsync(bsum, 0, 256 * 4, stream);
    hipMemsetAsync(bsq, 0, 256 * 4, stream);

    // conv2: K = 33*120 = 3960 (pad 3968), N = 210 (pad 224); 7 waves, 32 edges/block
    nnconv_msg_wmma<120, 210, 3968, 224, 2, 2>
        <<<N_EDGES / 32, 224, 0, stream>>>(h1, t2, W2T, ei, aggr);
    combine_kernel<120, 210><<<N_NODES, 224, 0, stream>>>(aggr, cnt, h1, root2, bias2, pre, bsum, bsq);
    bn_apply_kernel<210><<<(N_NODES * 210 + 255) / 256, 256, 0, stream>>>(pre, bsum, bsq, gamma2, beta2, h2);

    hipMemsetAsync(gsum, 0, (size_t)N_GRAPHS * H2_CH * 4, stream);
    hipMemsetAsync(gcnt, 0, (size_t)N_GRAPHS * 4, stream);
    pool_kernel<<<(N_NODES * H2_CH + 255) / 256, 256, 0, stream>>>(h2, batch, gsum, gcnt);
    head_kernel<<<N_GRAPHS, 64, 0, stream>>>(gsum, gcnt, lin1w, lin1b, lin2w, lin2b, out);
}
